// NystromNetPure_52810917872344
// MI455X (gfx1250) — matrix operand order repaired and checked by reading, not compiled, not run on hardware
//
#include <hip/hip_runtime.h>
#include <hip/hip_bf16.h>

typedef __attribute__((ext_vector_type(16))) __bf16       v16bf;
typedef __attribute__((ext_vector_type(8)))  float        v8f;
typedef __attribute__((ext_vector_type(4)))  unsigned int v4u;

namespace {
constexpr int kN    = 8192;   // rows of x
constexpr int kM    = 2048;   // samples / nystrom width
constexpr int kDin  = 256;    // input feature dim
constexpr int kDout = 100;    // classes

constexpr int kStageStride = 80;                 // bytes per staged B column (64 + 16 pad)
constexpr int kStageBytes  = 64 * kStageStride;  // one 32k x 64n bf16 panel

union Frag16 { v16bf h; v4u u[2]; };

// A fragment (16x32 bf16, row-major source [ld = K-stride]).
// lanes 0-15 hold M=0..15; half h=lane>>4 selects K sub-block:
//   VGPR0-3: K = h*16 + 0..7   VGPR4-7: K = h*16 + 8..15
__device__ __forceinline__ v16bf load_a_frag(const __bf16* __restrict__ base,
                                             int ld, int m0, int k0, int lane) {
  const int row  = m0 + (lane & 15);
  const int half = lane >> 4;
  const __bf16* p = base + (size_t)row * ld + k0 + half * 16;
  Frag16 f;
  f.u[0] = *(const v4u*)(p);
  f.u[1] = *(const v4u*)(p + 8);
  return f.h;
}

// B fragment (32x16 bf16) from W row-major [n][k] (B[k][n] = W[n][k]).
// N = lane&15, half = lane>>4, element e holds K = half*16 + e.
__device__ __forceinline__ v16bf load_b_frag(const __bf16* __restrict__ base,
                                             int ld, int k0, int n0, int lane) {
  const int col  = n0 + (lane & 15);
  const int half = lane >> 4;
  const __bf16* p = base + (size_t)col * ld + k0 + half * 16;
  Frag16 f;
  f.u[0] = *(const v4u*)(p);
  f.u[1] = *(const v4u*)(p + 8);
  return f.h;
}

__device__ __forceinline__ v16bf load_b_frag_guard(const __bf16* __restrict__ base,
                                                   int ld, int k0, int n0, int lane,
                                                   int nmax) {
  const int col  = n0 + (lane & 15);
  const int half = lane >> 4;
  Frag16 f = {};
  if (col < nmax) {
    const __bf16* p = base + (size_t)col * ld + k0 + half * 16;
    f.u[0] = *(const v4u*)(p);
    f.u[1] = *(const v4u*)(p + 8);
  }
  return f.h;
}

__device__ __forceinline__ v8f wmma_bf16(v16bf a, v16bf b, v8f c) {
  return __builtin_amdgcn_wmma_f32_16x16x32_bf16(false, a, false, b, (short)0, c,
                                                 false, false);
}

__device__ __forceinline__ float mish_f(float h) {
  const float sp = (h > 20.f) ? h : log1pf(__expf(h));
  return h * tanhf(sp);
}
}  // namespace

// ---- f32 -> bf16 conversion + row squared-norm (ncols fixed at 256) ----
__global__ void nys_rowsq_cvt(const float* __restrict__ in,
                              __bf16* __restrict__ out,
                              float* __restrict__ sq) {
  const int row = blockIdx.x;
  const int tid = threadIdx.x;                    // blockDim = 256 == kDin
  const float v = in[(size_t)row * kDin + tid];
  out[(size_t)row * kDin + tid] = (__bf16)v;
  __shared__ float red[256];
  red[tid] = v * v;
  __syncthreads();
  for (int s = 128; s > 0; s >>= 1) {
    if (tid < s) red[tid] += red[tid + s];
    __syncthreads();
  }
  if (tid == 0) sq[row] = red[0];
}

// ---- plain f32 -> bf16 ----
__global__ void nys_cvt_bf16(const float* __restrict__ in,
                             __bf16* __restrict__ out, int n) {
  const int i = blockIdx.x * blockDim.x + threadIdx.x;
  if (i < n) out[i] = (__bf16)in[i];
}

// ---- GEMM1: k = exp(-sqrt(max(x2+s2-2*x@s^T, eps)))  [8192 x 2048], K=256 ----
// block = 256 thr = 8 waves; wave tile = 16x64; block tile = 128x64.
__global__ __launch_bounds__(256) void nys_gemm_rbf(
    const __bf16* __restrict__ xbf, const __bf16* __restrict__ sbf,
    const float* __restrict__ xsq, const float* __restrict__ ssq,
    __bf16* __restrict__ kout) {
  const int lane = threadIdx.x & 31;
  const int wave = threadIdx.x >> 5;
  const int m0 = blockIdx.y * 128 + wave * 16;
  const int n0 = blockIdx.x * 64;
  v8f acc[4] = {};
#pragma unroll 2
  for (int k = 0; k < kDin; k += 32) {
    const v16bf a = load_a_frag(xbf, kDin, m0, k, lane);
#pragma unroll
    for (int t = 0; t < 4; ++t) {
      const v16bf b = load_b_frag(sbf, kDin, k, n0 + t * 16, lane);
      acc[t] = wmma_bf16(a, b, acc[t]);
    }
  }
  const int mbase = m0 + 8 * (lane >> 4);
  const int nlow  = n0 + (lane & 15);
#pragma unroll
  for (int t = 0; t < 4; ++t) {
    const int n = nlow + t * 16;
    const float sc = ssq[n];
#pragma unroll
    for (int r = 0; r < 8; ++r) {
      const int m = mbase + r;
      const float sq = xsq[m] + sc - 2.f * acc[t][r];
      const float d  = sqrtf(fmaxf(sq, 1e-12f));
      kout[(size_t)m * kM + n] = (__bf16)__expf(-d);
    }
  }
}

// ---- GEMM2: h = mish(k @ W_nys^T)  [8192 x 2048], K = 2048 (dominant) ----
// block = 256 thr = 8 waves; wave tile = 32x64 (2 A frags x 4 B frags = 8 WMMA
// per K-step); block tile = 256x64. B panel (32k x 64n) is staged in LDS via
// GLOBAL_LOAD_ASYNC_TO_LDS_B128 (ASYNCcnt), double-buffered, shared by all
// 8 waves -> 8x cut in B traffic from L2 and async copy overlapped with WMMA.
__global__ __launch_bounds__(256) void nys_gemm_mish(
    const __bf16* __restrict__ kbf, const __bf16* __restrict__ wbf,
    __bf16* __restrict__ hout) {
  __shared__ __attribute__((aligned(16))) unsigned char ldsB[2 * kStageBytes];
  const int tid  = threadIdx.x;
  const int lane = tid & 31;
  const int wave = tid >> 5;
  const int m0 = blockIdx.y * 256 + wave * 32;
  const int n0 = blockIdx.x * 64;

  // Per-thread staging slot: 256 threads x 16 B = 4 KB panel.
  const int sc = tid >> 2;   // column 0..63 of the panel
  const int sq = tid & 3;    // 16-byte chunk within the 64-byte column
  const unsigned ldsSlot = (unsigned)(sc * kStageStride + sq * 16);

  auto fill = [&](int stage, int k0) {
    const unsigned ldsOff =
        (unsigned)(uintptr_t)&ldsB[stage * kStageBytes] + ldsSlot;
    const unsigned gOff =
        (unsigned)(((unsigned)(n0 + sc) * (unsigned)kM + (unsigned)(k0 + sq * 8)) * 2u);
    // Wave-level async copy: 32 lanes x 16 B -> 512 B straight into LDS.
    asm volatile("global_load_async_to_lds_b128 %0, %1, %2"
                 :: "v"(ldsOff), "v"(gOff), "s"(wbf)
                 : "memory");
  };

  v8f acc[2][4] = {};
  fill(0, 0);
  for (int k = 0; k < kM; k += 32) {
    const int cur = (k >> 5) & 1;
    // Own wave's async fills complete, then all waves rendezvous: current
    // stage fully visible, all readers of the other stage are done with it.
    asm volatile("s_wait_asynccnt 0" ::: "memory");
    __syncthreads();
    if (k + 32 < kM) fill(cur ^ 1, k + 32);  // overlap next fill with compute

    const v16bf a0 = load_a_frag(kbf, kM, m0, k, lane);
    const v16bf a1 = load_a_frag(kbf, kM, m0 + 16, k, lane);
    const unsigned char* sb = &ldsB[cur * kStageBytes];
    const int bcol = (lane & 15);
    const int bhalf = lane >> 4;
#pragma unroll
    for (int t = 0; t < 4; ++t) {
      Frag16 f;
      const unsigned char* p =
          sb + (t * 16 + bcol) * kStageStride + bhalf * 32;
      f.u[0] = *(const v4u*)(p);
      f.u[1] = *(const v4u*)(p + 16);
      acc[0][t] = wmma_bf16(a0, f.h, acc[0][t]);
      acc[1][t] = wmma_bf16(a1, f.h, acc[1][t]);
    }
  }

  const int nlow = n0 + (lane & 15);
#pragma unroll
  for (int hm = 0; hm < 2; ++hm) {
    const int mbase = m0 + hm * 16 + 8 * (lane >> 4);
#pragma unroll
    for (int t = 0; t < 4; ++t) {
      const int n = nlow + t * 16;
#pragma unroll
      for (int r = 0; r < 8; ++r) {
        hout[(size_t)(mbase + r) * kM + n] = (__bf16)mish_f(acc[hm][t][r]);
      }
    }
  }
}

// ---- GEMM3: z = h @ W_p^T + b_p  [8192 x 100], K = 2048 ----
__global__ __launch_bounds__(256) void nys_gemm_proj(
    const __bf16* __restrict__ hbf, const __bf16* __restrict__ wpbf,
    const float* __restrict__ bp, float* __restrict__ z) {
  const int lane = threadIdx.x & 31;
  const int wave = threadIdx.x >> 5;
  const int m0 = blockIdx.y * 128 + wave * 16;
  const int n0 = blockIdx.x * 16;
  v8f acc = {};
#pragma unroll 2
  for (int k = 0; k < kM; k += 32) {
    const v16bf a = load_a_frag(hbf, kM, m0, k, lane);
    const v16bf b = load_b_frag_guard(wpbf, kM, k, n0, lane, kDout);
    acc = wmma_bf16(a, b, acc);   // EXEC reconverged: full mask here
  }
  const int n = n0 + (lane & 15);
  if (n < kDout) {
    const int mbase = m0 + 8 * (lane >> 4);
    const float bias = bp[n];
#pragma unroll
    for (int r = 0; r < 8; ++r)
      z[(size_t)(mbase + r) * kDout + n] = acc[r] + bias;
  }
}

// ---- log_softmax over rows of z [8192 x 100] ----
__global__ void nys_logsoftmax(const float* __restrict__ z,
                               float* __restrict__ out) {
  const int row = blockIdx.x;
  const int tid = threadIdx.x;                    // blockDim = 128
  __shared__ float red[128];
  float v = -3.4e38f;
  if (tid < kDout) v = z[(size_t)row * kDout + tid];
  red[tid] = v;
  __syncthreads();
  for (int s = 64; s > 0; s >>= 1) {
    if (tid < s) red[tid] = fmaxf(red[tid], red[tid + s]);
    __syncthreads();
  }
  const float mx = red[0];
  __syncthreads();
  red[tid] = (tid < kDout) ? __expf(v - mx) : 0.f;
  __syncthreads();
  for (int s = 64; s > 0; s >>= 1) {
    if (tid < s) red[tid] += red[tid + s];
    __syncthreads();
  }
  const float lse = __logf(red[0]);
  if (tid < kDout) out[(size_t)row * kDout + tid] = (v - mx) - lse;
}

extern "C" void kernel_launch(void* const* d_in, const int* in_sizes, int n_in,
                              void* d_out, int out_size, void* d_ws, size_t ws_size,
                              hipStream_t stream) {
  (void)in_sizes; (void)n_in; (void)out_size; (void)ws_size;
  const float* x       = (const float*)d_in[0];  // [8192, 256]
  const float* samples = (const float*)d_in[1];  // [2048, 256]
  const float* W_nys   = (const float*)d_in[2];  // [2048, 2048]
  const float* W_p     = (const float*)d_in[3];  // [100, 2048]
  const float* b_p     = (const float*)d_in[4];  // [100]
  float* out = (float*)d_out;                    // [8192, 100]

  char* ws = (char*)d_ws;
  auto take = [&](size_t bytes) -> char* {
    char* p = ws;
    ws += (bytes + 255) & ~(size_t)255;
    return p;
  };
  __bf16* x_bf  = (__bf16*)take((size_t)kN * kDin * 2);   //  4 MB
  __bf16* s_bf  = (__bf16*)take((size_t)kM * kDin * 2);   //  1 MB
  __bf16* wn_bf = (__bf16*)take((size_t)kM * kM * 2);     //  8 MB
  __bf16* wp_bf = (__bf16*)take((size_t)kDout * kM * 2);  //  0.4 MB
  float*  x_sq  = (float*)take((size_t)kN * 4);
  float*  s_sq  = (float*)take((size_t)kM * 4);
  __bf16* k_bf  = (__bf16*)take((size_t)kN * kM * 2);     // 32 MB
  __bf16* h_bf  = (__bf16*)take((size_t)kN * kM * 2);     // 32 MB
  float*  zbuf  = (float*)take((size_t)kN * kDout * 4);   //  3.2 MB

  nys_rowsq_cvt<<<kN, 256, 0, stream>>>(x, x_bf, x_sq);
  nys_rowsq_cvt<<<kM, 256, 0, stream>>>(samples, s_bf, s_sq);
  nys_cvt_bf16<<<(kM * kM + 255) / 256, 256, 0, stream>>>(W_nys, wn_bf, kM * kM);
  nys_cvt_bf16<<<(kDout * kM + 255) / 256, 256, 0, stream>>>(W_p, wp_bf, kDout * kM);

  nys_gemm_rbf<<<dim3(kM / 64, kN / 128), 256, 0, stream>>>(x_bf, s_bf, x_sq, s_sq, k_bf);
  nys_gemm_mish<<<dim3(kM / 64, kN / 256), 256, 0, stream>>>(k_bf, wn_bf, h_bf);
  nys_gemm_proj<<<dim3(7, kN / 128), 256, 0, stream>>>(h_bf, wp_bf, b_p, zbuf);
  nys_logsoftmax<<<kN, 128, 0, stream>>>(zbuf, out);
}
